// LSTM_8993661518513
// MI455X (gfx1250) — compile-verified, matrix operand order checked by reading, not run on hardware
//
#include <hip/hip_runtime.h>
#include <math.h>

// ---------------------------------------------------------------------------
// Fused LSTM (T=1024, B=256, I=H=128) + 18-class FC head for gfx1250 (CDNA5).
//
// v2 changes vs v1 (driven by disasm):
//  - W_hh B-fragments live in VGPRs for the whole T loop (no ds_load/wait on
//    the latency-critical recurrent WMMA chain).
//  - W_ih / FC B-fragments are loaded in batches of 4 before their WMMAs so a
//    single s_wait_dscnt covers them and the 4 independent gate chains issue
//    back-to-back.
//  - hbuf double-buffered -> one __syncthreads() per timestep instead of two.
// ---------------------------------------------------------------------------

typedef __attribute__((ext_vector_type(16))) __bf16 v16bf;
typedef __attribute__((ext_vector_type(8)))  float  v8f;
typedef __attribute__((ext_vector_type(4)))  float  v4f;

#define T_STEPS 1024
#define BATCH   256
#define I_DIM   128
#define H_DIM   128
#define NCLS    18
#define CHUNK   16      // batch rows per workgroup (one M tile)
#define NTHREADS 256    // 8 waves (wave32)

// LDS layout (element offsets, __bf16 units):
//  wih frags : 32 ntiles * 4 ktiles * 512 = 65536   (131072 B)
//  whh frags : 65536 (staging only; copied to VGPRs) (131072 B)
//  fc  frags : 2 * 4 * 512 = 4096                   (  8192 B)
//  hbuf[2]   : 2*16*128 = 4096                      (  8192 B)
//  xbuf[2]   : 2*2048 = 4096                        (  8192 B)
#define OFF_WIH 0
#define OFF_WHH 65536
#define OFF_FC  131072
#define OFF_H   135168
#define OFF_X   139264
#define SMEM_ELEMS 143360
#define SMEM_BYTES (SMEM_ELEMS * 2)   // 286720 B < 320KB/WGP

union AFrag { v16bf v; v4f f4[2]; };

// K index within a 32-wide K tile for fragment element e (v=e>>1, pair j=e&1)
// per the CDNA5 16-bit A-matrix 16x32 layout (lanes 0-15: K 0..7,16..23;
// lanes 16-31: K 8..15,24..31).
__device__ __forceinline__ int kidx(int e, int khalf) {
    int v = e >> 1, j = e & 1;
    return ((v >= 4) ? 16 : 0) + khalf * 8 + (v & 3) * 2 + j;
}

// Convert a row-major f32 weight [nrows x 128] into bf16 B-fragments.
// Fragment (nt,kt): 32 lanes * 16 bf16, stored lane-contiguous (32B/lane).
// B[k][n] = W[nt*16 + n][kt*32 + k-in-tile]; rows >= nrows are zero padding.
__device__ void fill_wfrag(__bf16* dst, const float* __restrict__ W,
                           int ntiles, int nrows, int tid) {
    int total = ntiles * 4 * 32 * 16;
    for (int idx = tid; idx < total; idx += NTHREADS) {
        int e     = idx & 15;
        int lane  = (idx >> 4) & 31;
        int kt    = (idx >> 9) & 3;
        int nt    = idx >> 11;
        int khalf = lane >> 4;
        int n     = nt * 16 + (lane & 15);
        int k     = kt * 32 + kidx(e, khalf);
        float val = (n < nrows) ? W[n * 128 + k] : 0.0f;
        dst[(nt * 4 + kt) * 512 + lane * 16 + e] = (__bf16)val;
    }
}

// Load an A fragment (16x32 K-tile kt) from a row-major bf16 [16][128] buffer.
// Two 16B LDS reads per lane (ds_load_b128).
__device__ __forceinline__ v16bf load_afrag(const __bf16* buf, int kt, int lane) {
    int m = lane & 15, khalf = lane >> 4;
    int kbase = kt * 32 + khalf * 8;
    AFrag a;
    a.f4[0] = *(const v4f*)(buf + m * 128 + kbase);
    a.f4[1] = *(const v4f*)(buf + m * 128 + kbase + 16);
    return a.v;
}

__device__ __forceinline__ v16bf load_bfrag(const __bf16* frags, int fragIdx, int lane) {
    const __bf16* p = frags + fragIdx * 512 + lane * 16;
    AFrag b;
    b.f4[0] = *(const v4f*)(p);
    b.f4[1] = *(const v4f*)(p + 8);
    return b.v;
}

__device__ __forceinline__ v8f wmma_bf16(v16bf a, v16bf b, v8f c) {
    return __builtin_amdgcn_wmma_f32_16x16x32_bf16(false, a, false, b,
                                                   (short)0, c, false, false);
}

__device__ __forceinline__ float sigmoidf_fast(float x) {
    return 1.0f / (1.0f + __expf(-x));
}

extern "C" __global__ void __launch_bounds__(NTHREADS, 1)
lstm_fused(const float* __restrict__ x,     // [T,B,I]
           const float* __restrict__ h0,    // [B,H]
           const float* __restrict__ c0,    // [B,H]
           const float* __restrict__ W_ih,  // [512,128]
           const float* __restrict__ W_hh,  // [512,128]
           const float* __restrict__ b_ih,  // [512]
           const float* __restrict__ b_hh,  // [512]
           const float* __restrict__ fc_W,  // [18,128]
           const float* __restrict__ fc_b,  // [18]
           float* __restrict__ out)         // [T*B*18] ++ hT[B*H] ++ cT[B*H]
{
    extern __shared__ __bf16 smem[];
    __bf16* wih  = smem + OFF_WIH;
    __bf16* whh  = smem + OFF_WHH;
    __bf16* fcf  = smem + OFF_FC;
    __bf16* hbuf = smem + OFF_H;    // double-buffered [2][16][128]
    __bf16* xbuf = smem + OFF_X;    // double-buffered [2][16][128]

    const int tid   = threadIdx.x;
    const int lane  = tid & 31;
    const int w     = tid >> 5;       // wave id 0..7 -> hidden cols 16w..16w+15
    const int b0    = blockIdx.x * CHUNK;
    const int m     = lane & 15;
    const int khalf = lane >> 4;

    // ---- one-time setup: weights -> LDS fragments, h0 -> hbuf[0], x0 -> xbuf[0]
    fill_wfrag(wih, W_ih, 32, 512, tid);
    fill_wfrag(whh, W_hh, 32, 512, tid);
    fill_wfrag(fcf, fc_W, 2, NCLS, tid);

    for (int idx = tid; idx < CHUNK * H_DIM; idx += NTHREADS)
        hbuf[idx] = (__bf16)h0[(size_t)(b0 + idx / H_DIM) * H_DIM + (idx % H_DIM)];

    {
        int row = tid >> 4, col = (tid & 15) * 8;
        const float* src = x + (size_t)(b0 + row) * I_DIM + col;
        #pragma unroll
        for (int e = 0; e < 8; e++) xbuf[row * 128 + col + e] = (__bf16)src[e];
    }
    __syncthreads();

    // ---- W_hh B-fragments: resident in VGPRs for the whole T loop ----------
    v16bf whhr[4][4];   // [gate][ktile] : 128 VGPRs
    #pragma unroll
    for (int g = 0; g < 4; g++)
        #pragma unroll
        for (int kt = 0; kt < 4; kt++)
            whhr[g][kt] = load_bfrag(whh, (g * 8 + w) * 4 + kt, lane);

    // ---- per-lane persistent state
    float bias[4];
    #pragma unroll
    for (int g = 0; g < 4; g++) {
        int col = g * 128 + w * 16 + m;           // column within 4H
        bias[g] = b_ih[col] + b_hh[col];
    }
    float fcb = 0.0f;
    {
        int col = w * 16 + m;
        if (w < 2 && col < NCLS) fcb = fc_b[col];
    }
    v8f c;                                        // cell state, lives in VGPRs
    #pragma unroll
    for (int r = 0; r < 8; r++) {
        int mr = r + 8 * khalf;
        c[r] = c0[(size_t)(b0 + mr) * H_DIM + w * 16 + m];
    }

    float* hT = out + (size_t)T_STEPS * BATCH * NCLS;
    float* cT = hT + (size_t)BATCH * H_DIM;

    float xr[8];
    for (int t = 0; t < T_STEPS; t++) {
        const int par = t & 1;

        // prefetch x_{t+1} into registers; wait lands at the LDS write below
        if (t + 1 < T_STEPS) {
            int row = tid >> 4, col = (tid & 15) * 8;
            const float* src = x + (size_t)(t + 1) * BATCH * I_DIM
                                 + (size_t)(b0 + row) * I_DIM + col;
            #pragma unroll
            for (int e = 0; e < 8; e++) xr[e] = src[e];
        }

        // A fragments of h_{t-1} (shared by recurrence and fused FC head)
        v16bf ah[4];
        #pragma unroll
        for (int kt = 0; kt < 4; kt++)
            ah[kt] = load_afrag(hbuf + par * 2048, kt, lane);

        // fused FC head: out[t-1] = h_{t-1} @ fc_W^T + fc_b (waves 0,1)
        if (w < 2 && t > 0) {
            v16bf bwf[4];
            #pragma unroll
            for (int kt = 0; kt < 4; kt++)
                bwf[kt] = load_bfrag(fcf, w * 4 + kt, lane);
            v8f facc;
            #pragma unroll
            for (int r = 0; r < 8; r++) facc[r] = fcb;
            #pragma unroll
            for (int kt = 0; kt < 4; kt++)
                facc = wmma_bf16(ah[kt], bwf[kt], facc);
            int gcol = w * 16 + m;
            if (gcol < NCLS) {
                #pragma unroll
                for (int r = 0; r < 8; r++) {
                    int mr = r + 8 * khalf;
                    out[(size_t)(t - 1) * BATCH * NCLS
                        + (size_t)(b0 + mr) * NCLS + gcol] = facc[r];
                }
            }
        }

        // gates[16,512] = x_t @ W_ih^T + h_{t-1} @ W_hh^T + b
        v8f acc[4];
        #pragma unroll
        for (int g = 0; g < 4; g++)
            #pragma unroll
            for (int r = 0; r < 8; r++) acc[g][r] = bias[g];

        // x half: B from LDS, batched 4-at-a-time (one dscnt wait per ktile,
        // then 4 independent accumulator chains issue back-to-back)
        #pragma unroll
        for (int kt = 0; kt < 4; kt++) {
            v16bf a = load_afrag(xbuf + par * 2048, kt, lane);
            v16bf bw[4];
            #pragma unroll
            for (int g = 0; g < 4; g++)
                bw[g] = load_bfrag(wih, (g * 8 + w) * 4 + kt, lane);
            #pragma unroll
            for (int g = 0; g < 4; g++)
                acc[g] = wmma_bf16(a, bw[g], acc[g]);
        }
        // h half: B resident in VGPRs -> no LDS traffic on the critical path
        #pragma unroll
        for (int kt = 0; kt < 4; kt++) {
            #pragma unroll
            for (int g = 0; g < 4; g++)
                acc[g] = wmma_bf16(ah[kt], whhr[g][kt], acc[g]);
        }

        // elementwise cell update (register-local: wave w owns cols 16w..16w+15)
        __bf16 hb[8];
        float  hf[8];
        #pragma unroll
        for (int r = 0; r < 8; r++) {
            float ig = sigmoidf_fast(acc[0][r]);
            float fg = sigmoidf_fast(acc[1][r]);
            float gg = tanhf(acc[2][r]);
            float og = sigmoidf_fast(acc[3][r]);
            float cn = fg * c[r] + ig * gg;
            c[r] = cn;
            float hn = og * tanhf(cn);
            hf[r] = hn;
            hb[r] = (__bf16)hn;
        }

        // writes go to the *other* buffers (last read in t-1, fenced by t-1's
        // barrier), so a single end-of-iteration barrier suffices
        #pragma unroll
        for (int r = 0; r < 8; r++) {
            int mr = r + 8 * khalf;
            hbuf[(1 - par) * 2048 + mr * 128 + w * 16 + m] = hb[r];
        }
        if (t + 1 < T_STEPS) {
            int row = tid >> 4, col = (tid & 15) * 8;
            __bf16* dst = xbuf + (1 - par) * 2048 + row * 128 + col;
            #pragma unroll
            for (int e = 0; e < 8; e++) dst[e] = (__bf16)xr[e];
        }
        if (t == T_STEPS - 1) {
            #pragma unroll
            for (int r = 0; r < 8; r++) {
                int mr = r + 8 * khalf;
                hT[(size_t)(b0 + mr) * H_DIM + w * 16 + m] = hf[r];
                cT[(size_t)(b0 + mr) * H_DIM + w * 16 + m] = c[r];
            }
        }
        __syncthreads();
    }

    // FC head for the final step's hidden state (last h written to buffer 0)
    if (w < 2) {
        v16bf bwf[4];
        #pragma unroll
        for (int kt = 0; kt < 4; kt++)
            bwf[kt] = load_bfrag(fcf, w * 4 + kt, lane);
        v8f facc;
        #pragma unroll
        for (int r = 0; r < 8; r++) facc[r] = fcb;
        #pragma unroll
        for (int kt = 0; kt < 4; kt++)
            facc = wmma_bf16(load_afrag(hbuf, kt, lane), bwf[kt], facc);
        int gcol = w * 16 + m;
        if (gcol < NCLS) {
            #pragma unroll
            for (int r = 0; r < 8; r++) {
                int mr = r + 8 * khalf;
                out[(size_t)(T_STEPS - 1) * BATCH * NCLS
                    + (size_t)(b0 + mr) * NCLS + gcol] = facc[r];
            }
        }
    }
}

extern "C" void kernel_launch(void* const* d_in, const int* in_sizes, int n_in,
                              void* d_out, int out_size, void* d_ws, size_t ws_size,
                              hipStream_t stream) {
    const float* x    = (const float*)d_in[0];
    const float* h0   = (const float*)d_in[1];  // [1,B,H]
    const float* c0   = (const float*)d_in[2];  // [1,B,H]
    const float* W_ih = (const float*)d_in[3];
    const float* W_hh = (const float*)d_in[4];
    const float* b_ih = (const float*)d_in[5];
    const float* b_hh = (const float*)d_in[6];
    const float* fc_W = (const float*)d_in[7];
    const float* fc_b = (const float*)d_in[8];
    float* out = (float*)d_out;

    // gfx1250 supports up to 320KB LDS per workgroup; we use ~280KB.
    (void)hipFuncSetAttribute((const void*)lstm_fused,
                              hipFuncAttributeMaxDynamicSharedMemorySize,
                              SMEM_BYTES);

    lstm_fused<<<dim3(BATCH / CHUNK), dim3(NTHREADS), SMEM_BYTES, stream>>>(
        x, h0, c0, W_ih, W_hh, b_ih, b_hh, fc_W, fc_b, out);
}
